// VQBehaviorTransformer_81423989998115
// MI455X (gfx1250) — compile-verified, matrix operand order checked by reading, not run on hardware
//
#include <hip/hip_runtime.h>
#include <hip/hip_bf16.h>

// ---------------- problem constants ----------------
#define NTOK 1280
#define EDIM 1024
#define HID  1024
#define GG   2
#define CC   512
#define DD   64
#define WAW  70          // W*A
#define GCC  1024        // G*C
#define O3N  71680       // G*C*WA
#define OUT_LOSS_IDX 89600

typedef __attribute__((ext_vector_type(16))) __bf16 v16bf;
typedef __attribute__((ext_vector_type(8)))  float  v8f;

__device__ __forceinline__ v8f wmma_bf16(v16bf a, v16bf b, v8f c)
{
    return __builtin_amdgcn_wmma_f32_16x16x32_bf16(false, a, false, b,
                                                   (short)0, c, false, false);
}

// =====================================================================
// Pack fp32 row-major (M x K) -> bf16 WMMA A-tile order.
// Tile (mt,kt) = 16(M) x 32(K); within tile: lane L = (m&15) + 16*half,
// lane holds 16 bf16: v=2j+p -> k_local = (j<4 ? 2j : 16+2(j-4)) + 8*half + p
// =====================================================================
__global__ __launch_bounds__(256) void pack_a_bf16(const float* __restrict__ src,
                                                   __bf16* __restrict__ dst,
                                                   int M, int K)
{
    int tid = blockIdx.x * blockDim.x + threadIdx.x;
    int total = (M * K) >> 1;                 // one dword (2 bf16) per thread
    if (tid >= total) return;
    int tile   = tid >> 8;                    // 256 pairs per tile
    int within = tid & 255;
    int lane   = within >> 3;
    int j      = within & 7;
    int Kt = K >> 5;
    int mt = tile / Kt, kt = tile % Kt;
    int half = lane >> 4;
    int kb = (j < 4 ? 2 * j : 16 + 2 * (j - 4)) + half * 8;
    int m = mt * 16 + (lane & 15);
    int k = kt * 32 + kb;
    float f0 = src[(size_t)m * K + k];
    float f1 = src[(size_t)m * K + k + 1];
    __bf16 b0 = (__bf16)f0, b1 = (__bf16)f1;
    unsigned int w = ((unsigned int)__builtin_bit_cast(unsigned short, b1) << 16)
                   |  (unsigned int)__builtin_bit_cast(unsigned short, b0);
    ((unsigned int*)dst)[tid] = w;
}

// Pack fp32 row-major weight (K x N) -> bf16 WMMA B-tile order.
__global__ __launch_bounds__(256) void pack_b_bf16(const float* __restrict__ src,
                                                   __bf16* __restrict__ dst,
                                                   int K, int N)
{
    int tid = blockIdx.x * blockDim.x + threadIdx.x;
    int total = (K * N) >> 1;
    if (tid >= total) return;
    int tile   = tid >> 8;
    int within = tid & 255;
    int lane   = within >> 3;
    int j      = within & 7;
    int Nt = N >> 4;
    int kt = tile / Nt, nt = tile % Nt;
    int half = lane >> 4;
    int kb = (j < 4 ? 2 * j : 16 + 2 * (j - 4)) + half * 8;
    int n = nt * 16 + (lane & 15);
    int k = kt * 32 + kb;
    float f0 = src[(size_t)k * N + n];
    float f1 = src[(size_t)(k + 1) * N + n];
    __bf16 b0 = (__bf16)f0, b1 = (__bf16)f1;
    unsigned int w = ((unsigned int)__builtin_bit_cast(unsigned short, b1) << 16)
                   |  (unsigned int)__builtin_bit_cast(unsigned short, b0);
    ((unsigned int*)dst)[tid] = w;
}

// =====================================================================
// bf16 WMMA GEMM: C(M x N) = act(A(M x K) * B(K x N) + bias)
// Each wave owns a 32(M) x 64(N) block: 8 accumulators, 8 WMMAs per
// K-tile, with double-buffered operand loads (K unrolled by 2) so the
// next tile's 12 b128 loads are in flight while the current 8 WMMAs run.
// =====================================================================
__global__ __launch_bounds__(256) void gemm_bf16_wmma(const __bf16* __restrict__ Ap,
                                                      const __bf16* __restrict__ Bp,
                                                      const float* __restrict__ bias,
                                                      float* __restrict__ Cout,
                                                      int Mtiles2, int Ktiles,
                                                      int Nstrips, int N, int relu)
{
    int wave = (blockIdx.x * blockDim.x + threadIdx.x) >> 5;
    int lane = threadIdx.x & 31;
    int mt2 = wave / Nstrips;          // pair-of-16-row tiles
    int ns  = wave % Nstrips;
    if (mt2 >= Mtiles2) return;        // uniform per wave
    int n0t  = ns * 4;                 // first 16-col tile of this strip
    int Nt16 = N >> 4;
    int mt0  = mt2 * 2;

    const __bf16* aBase0 = Ap + ((size_t)(mt0 * Ktiles) << 9) + lane * 16;
    const __bf16* aBase1 = aBase0 + ((size_t)Ktiles << 9);
    const __bf16* bBase  = Bp + ((size_t)n0t << 9) + lane * 16;
    const size_t  bStride = (size_t)Nt16 << 9;

    v8f acc0 = {}, acc1 = {}, acc2 = {}, acc3 = {};
    v8f acc4 = {}, acc5 = {}, acc6 = {}, acc7 = {};

    v16bf xa0, xa1, xb0, xb1, xb2, xb3;    // buffer X
    v16bf ya0, ya1, yb0, yb1, yb2, yb3;    // buffer Y

#define LOAD_SET(A0, A1, B0, B1, B2, B3, KT)                                  \
    do {                                                                      \
        const __bf16* bp_ = bBase + (size_t)(KT) * bStride;                   \
        A0 = *(const v16bf*)(aBase0 + ((size_t)(KT) << 9));                   \
        A1 = *(const v16bf*)(aBase1 + ((size_t)(KT) << 9));                   \
        B0 = *(const v16bf*)(bp_);                                            \
        B1 = *(const v16bf*)(bp_ + 512);                                      \
        B2 = *(const v16bf*)(bp_ + 1024);                                     \
        B3 = *(const v16bf*)(bp_ + 1536);                                     \
    } while (0)

#define WMMA_SET(A0, A1, B0, B1, B2, B3)                                      \
    do {                                                                      \
        acc0 = wmma_bf16(A0, B0, acc0);                                       \
        acc1 = wmma_bf16(A0, B1, acc1);                                       \
        acc2 = wmma_bf16(A0, B2, acc2);                                       \
        acc3 = wmma_bf16(A0, B3, acc3);                                       \
        acc4 = wmma_bf16(A1, B0, acc4);                                       \
        acc5 = wmma_bf16(A1, B1, acc5);                                       \
        acc6 = wmma_bf16(A1, B2, acc6);                                       \
        acc7 = wmma_bf16(A1, B3, acc7);                                       \
    } while (0)

    LOAD_SET(xa0, xa1, xb0, xb1, xb2, xb3, 0);
    for (int kt = 0; kt < Ktiles; kt += 2) {   // Ktiles must be even (it is: 32)
        LOAD_SET(ya0, ya1, yb0, yb1, yb2, yb3, kt + 1);
        WMMA_SET(xa0, xa1, xb0, xb1, xb2, xb3);
        if (kt + 2 < Ktiles)
            LOAD_SET(xa0, xa1, xb0, xb1, xb2, xb3, kt + 2);
        WMMA_SET(ya0, ya1, yb0, yb1, yb2, yb3);
    }
#undef LOAD_SET
#undef WMMA_SET

    // Epilogue. C/D layout: n = lane&15, m = r + 8*(lane>>4).
    int half = lane >> 4;
    int ncol = lane & 15;
    v8f accs[8] = {acc0, acc1, acc2, acc3, acc4, acc5, acc6, acc7};
    #pragma unroll
    for (int h = 0; h < 2; ++h) {
        int mrow0 = (mt0 + h) * 16 + 8 * half;
        #pragma unroll
        for (int nt = 0; nt < 4; ++nt) {
            int n = (n0t + nt) * 16 + ncol;
            float bv = bias[n];
            #pragma unroll
            for (int r = 0; r < 8; ++r) {
                float v = accs[h * 4 + nt][r] + bv;
                if (relu) v = fmaxf(v, 0.0f);
                Cout[(size_t)(mrow0 + r) * N + n] = v;
            }
        }
    }
}

// =====================================================================
// Gumbel-max categorical sample + focal loss. One wave per (token, group).
// =====================================================================
__device__ __forceinline__ unsigned int hash_u32(unsigned int x)
{
    x ^= x >> 16; x *= 0x7feb352dU;
    x ^= x >> 15; x *= 0x846ca68bU;
    x ^= x >> 16; return x;
}

__global__ __launch_bounds__(256) void sample_focal(const float* __restrict__ logits,
                                                    const int* __restrict__ bins,
                                                    int* __restrict__ centers,
                                                    float* __restrict__ acc)
{
    int wid  = (blockIdx.x * blockDim.x + threadIdx.x) >> 5;
    int lane = threadIdx.x & 31;
    int t = wid >> 1, g = wid & 1;
    if (t >= NTOK) return;
    const float* base = logits + (size_t)t * GCC + g * CC;

    float zbest = -__builtin_inff(); int ibest = 0;
    float lmax  = -__builtin_inff();
    #pragma unroll 4
    for (int s = 0; s < 16; ++s) {
        int i = lane + (s << 5);
        float l = base[i];
        lmax = fmaxf(lmax, l);
        unsigned int h = hash_u32((unsigned int)(t * 977 + g * 499 + i) ^ 0x2545F491u);
        float u = ((float)(h >> 8) + 0.5f) * (1.0f / 16777216.0f);
        float z = l - __logf(-__logf(u));      // TEMP == 1
        if (z > zbest) { zbest = z; ibest = i; }
    }
    #pragma unroll
    for (int off = 16; off > 0; off >>= 1) {
        float oz = __shfl_xor(zbest, off, 32);
        int   oi = __shfl_xor(ibest, off, 32);
        if (oz > zbest || (oz == zbest && oi < ibest)) { zbest = oz; ibest = oi; }
        lmax = fmaxf(lmax, __shfl_xor(lmax, off, 32));
    }
    float se = 0.0f;
    #pragma unroll 4
    for (int s = 0; s < 16; ++s) se += __expf(base[lane + (s << 5)] - lmax);
    #pragma unroll
    for (int off = 16; off > 0; off >>= 1) se += __shfl_xor(se, off, 32);

    if (lane == 0) {
        centers[t * 2 + g] = ibest;
        int tgt = bins[t * 2 + g];
        float lp = (base[tgt] - lmax) - __logf(se);
        float pt = __expf(lp);
        float om = 1.0f - pt;
        atomicAdd(&acc[g], -om * om * lp);        // GAMMA = 2
    }
}

// =====================================================================
// Per-token: codebook decode (Wd) + gathered O3 offset GEMV + |.| loss.
// 140 threads own (g,j) columns of O3 (contiguous 70-float runs per
// k-row -> coalesced). global_prefetch_b8 issued 16 rows ahead to keep
// the 286KB-strided row stream resident (memory-bound portion).
// =====================================================================
__global__ __launch_bounds__(256) void decode_gather(const float* __restrict__ oh2,
                                                     const float* __restrict__ O3,
                                                     const float* __restrict__ ob3,
                                                     const float* __restrict__ cb,
                                                     const float* __restrict__ Wd,
                                                     const float* __restrict__ bd,
                                                     const int* __restrict__ centers,
                                                     const float* __restrict__ aseq,
                                                     float* __restrict__ out,
                                                     float* __restrict__ acc)
{
    __shared__ float sh_oh[HID];
    __shared__ float sh_cent[GG * DD];
    __shared__ float sh_off[GG * WAW];
    __shared__ float sred[256];

    int t = blockIdx.x, tid = threadIdx.x;
    for (int i = tid; i < HID; i += 256) sh_oh[i] = oh2[(size_t)t * HID + i];
    int c0 = centers[t * 2], c1 = centers[t * 2 + 1];
    if (tid < GG * DD) {
        int g = tid >> 6, d = tid & 63;
        int c = g ? c1 : c0;
        sh_cent[tid] = cb[((size_t)g * CC + c) * DD + d];
    }
    __syncthreads();

    if (tid < GG * WAW) {
        int g = tid / WAW, j = tid - g * WAW;
        int c = g ? c1 : c0;
        size_t colbase = (size_t)g * (CC * WAW) + (size_t)c * WAW + j;
        float s = ob3[colbase];
        const float* p = O3 + colbase;
        #pragma unroll 4
        for (int k = 0; k < HID; ++k) {
            int kp = (k + 16 < HID) ? (k + 16) : k;     // clamped look-ahead
            __builtin_prefetch(p + (size_t)kp * O3N, 0, 0);
            s += sh_oh[k] * p[(size_t)k * O3N];
        }
        sh_off[tid] = s;
    }
    __syncthreads();

    float la = 0.0f;
    if (tid < WAW) {
        int j = tid;
        float dec = bd[j];
        #pragma unroll 4
        for (int k = 0; k < GG * DD; ++k) dec += sh_cent[k] * Wd[k * WAW + j];
        float ah = dec + sh_off[j] + sh_off[WAW + j];
        out[(size_t)t * WAW + j] = ah;
        la = fabsf(aseq[(size_t)t * WAW + j] - ah);
    }
    sred[tid] = la;
    __syncthreads();
    #pragma unroll
    for (int s = 128; s > 0; s >>= 1) {
        if (tid < s) sred[tid] += sred[tid + s];
        __syncthreads();
    }
    if (tid == 0) atomicAdd(&acc[2], sred[0]);
}

__global__ void init_acc(float* acc) { if (threadIdx.x < 4) acc[threadIdx.x] = 0.0f; }

__global__ void finalize_loss(const float* __restrict__ acc, float* __restrict__ out)
{
    // cbet = 5*focal0 + 0.5*focal1 (means over NT); + 100 * mean|offset err|
    out[OUT_LOSS_IDX] = 5.0f * acc[0] / (float)NTOK
                      + 0.5f * acc[1] / (float)NTOK
                      + 100.0f * (acc[2] / ((float)NTOK * (float)WAW));
}

// =====================================================================
extern "C" void kernel_launch(void* const* d_in, const int* in_sizes, int n_in,
                              void* d_out, int out_size, void* d_ws, size_t ws_size,
                              hipStream_t stream)
{
    (void)in_sizes; (void)n_in; (void)out_size; (void)ws_size;
    const float* X   = (const float*)d_in[0];
    const float* aseq= (const float*)d_in[1];
    const int*   bins= (const int*)  d_in[2];
    const float* W1  = (const float*)d_in[3];  const float* b1  = (const float*)d_in[4];
    const float* W2  = (const float*)d_in[5];  const float* b2  = (const float*)d_in[6];
    const float* W3  = (const float*)d_in[7];  const float* b3  = (const float*)d_in[8];
    const float* O1  = (const float*)d_in[9];  const float* ob1 = (const float*)d_in[10];
    const float* O2  = (const float*)d_in[11]; const float* ob2 = (const float*)d_in[12];
    const float* O3  = (const float*)d_in[13]; const float* ob3 = (const float*)d_in[14];
    const float* cb  = (const float*)d_in[15];
    const float* Wd  = (const float*)d_in[16]; const float* bd  = (const float*)d_in[17];
    float* out = (float*)d_out;

    char* ws = (char*)d_ws;
    size_t off = 0;
    auto take = [&](size_t bytes) -> char* {
        char* p = ws + off;
        off = (off + bytes + 255) & ~(size_t)255;
        return p;
    };
    __bf16* Xp   = (__bf16*)take((size_t)NTOK * EDIM * 2);
    __bf16* W1p  = (__bf16*)take((size_t)EDIM * HID * 2);
    __bf16* W2p  = (__bf16*)take((size_t)HID * HID * 2);
    __bf16* W3p  = (__bf16*)take((size_t)HID * GCC * 2);
    __bf16* O1p  = (__bf16*)take((size_t)EDIM * HID * 2);
    __bf16* O2p  = (__bf16*)take((size_t)HID * HID * 2);
    __bf16* actA = (__bf16*)take((size_t)NTOK * HID * 2);
    __bf16* actB = (__bf16*)take((size_t)NTOK * HID * 2);
    float*  tmpf    = (float*)take((size_t)NTOK * HID * 4);
    float*  logitsf = (float*)take((size_t)NTOK * GCC * 4);
    float*  oh2f    = (float*)take((size_t)NTOK * HID * 4);
    int*    centers = (int*)  take((size_t)NTOK * GG * 4);
    float*  acc     = (float*)take(16 * 4);

    const int TB = 256;
    const int packA_blk = (NTOK * EDIM / 2 + TB - 1) / TB;   // 2560
    const int packB_blk = (EDIM * HID  / 2 + TB - 1) / TB;   // 2048
    const int Mt2 = NTOK / 32, Kt = EDIM / 32, Nstrips = HID / 64;
    const int gemm_blk = (Mt2 * Nstrips) / 8;                // 80 blocks of 8 waves

    init_acc<<<1, 32, 0, stream>>>(acc);

    // one-time-per-call bf16 packing (memory passes, ~39 MB total)
    pack_a_bf16<<<packA_blk, TB, 0, stream>>>(X,  Xp,  NTOK, EDIM);
    pack_b_bf16<<<packB_blk, TB, 0, stream>>>(W1, W1p, EDIM, HID);
    pack_b_bf16<<<packB_blk, TB, 0, stream>>>(W2, W2p, HID,  HID);
    pack_b_bf16<<<packB_blk, TB, 0, stream>>>(W3, W3p, HID,  GCC);
    pack_b_bf16<<<packB_blk, TB, 0, stream>>>(O1, O1p, EDIM, HID);
    pack_b_bf16<<<packB_blk, TB, 0, stream>>>(O2, O2p, HID,  HID);

    // logits chain
    gemm_bf16_wmma<<<gemm_blk, TB, 0, stream>>>(Xp,   W1p, b1, tmpf,    Mt2, Kt, Nstrips, HID, 1);
    pack_a_bf16  <<<packA_blk, TB, 0, stream>>>(tmpf, actA, NTOK, HID);
    gemm_bf16_wmma<<<gemm_blk, TB, 0, stream>>>(actA, W2p, b2, tmpf,    Mt2, Kt, Nstrips, HID, 1);
    pack_a_bf16  <<<packA_blk, TB, 0, stream>>>(tmpf, actB, NTOK, HID);
    gemm_bf16_wmma<<<gemm_blk, TB, 0, stream>>>(actB, W3p, b3, logitsf, Mt2, Kt, Nstrips, GCC, 0);

    // offset-MLP hidden chain (O3 handled as a gather-GEMV later)
    gemm_bf16_wmma<<<gemm_blk, TB, 0, stream>>>(Xp,   O1p, ob1, tmpf,   Mt2, Kt, Nstrips, HID, 1);
    pack_a_bf16  <<<packA_blk, TB, 0, stream>>>(tmpf, actA, NTOK, HID);
    gemm_bf16_wmma<<<gemm_blk, TB, 0, stream>>>(actA, O2p, ob2, oh2f,   Mt2, Kt, Nstrips, HID, 1);

    // sample centers + focal loss: 2560 waves
    sample_focal<<<(NTOK * GG) / 8, TB, 0, stream>>>(logitsf, bins, centers, acc);

    // decode + gathered offsets + L1 loss; writes a_hat to d_out
    decode_gather<<<NTOK, TB, 0, stream>>>(oh2f, O3, ob3, cb, Wd, bd, centers,
                                           aseq, out, acc);

    finalize_loss<<<1, 1, 0, stream>>>(acc, out);
}
